// DeformableTransformerDecoderLayer_52913997087415
// MI455X (gfx1250) — matrix operand order repaired and checked by reading, not compiled, not run on hardware
//
#include <hip/hip_runtime.h>
#include <hip/hip_bf16.h>

typedef __attribute__((ext_vector_type(16))) _Float16     v16h;
typedef __attribute__((ext_vector_type(8)))  float        v8f;
typedef __attribute__((ext_vector_type(4)))  unsigned int v4u;
typedef __attribute__((ext_vector_type(8)))  int          v8i;
typedef __attribute__((ext_vector_type(4)))  int          v4i;

#define B_    8
#define LQ_   1800
#define D_    256
#define H_    8
#define DH_   32
#define DFF_  1024
#define LIN_  19947
#define NROW  (B_*LQ_)      // 14400
#define VROW  (B_*LIN_)     // 159576

#if defined(__has_builtin)
#if __has_builtin(__builtin_amdgcn_tensor_load_to_lds)
#define HAVE_TDM 1
#endif
#endif

#ifdef HAVE_TDM
// ---------------------------------------------------------------------------
// Tensor Data Mover: 2D tile load (rows x 32 halfs, row stride 256 halfs)
// global -> LDS, with TDM-side padding of +8 halfs after every 32 halfs so
// the DMA directly produces the bank-spread 40-half LDS row layout.
// D# per CDNA5 ISA. Rows >= rows_remaining zero-fill.
// ---------------------------------------------------------------------------
__device__ __forceinline__ void tdm_load_tile32(const _Float16* gsrc,
                                                unsigned lds_off,
                                                int rows_remaining) {
    unsigned long long ga = (unsigned long long)(const void*)gsrc;
    v4u g0;
    g0.x = 1u;                                        // count=1 (valid), user mode
    g0.y = lds_off;                                   // lds_addr (bytes)
    g0.z = (unsigned)(ga & 0xFFFFFFFFu);              // global_addr[31:0]
    g0.w = (unsigned)((ga >> 32) & 0x01FFFFFFu)       // global_addr[56:32]
         | 0x80000000u;                               // type = 2 ("image")
    v8i g1;
    g1[0] = (1 << 16)      // data_size = 2 bytes
          | (1 << 20)      // pad_enable
          | (3 << 22)      // pad_interval: 16 DWORDs (= 32 halfs)
          | (3 << 25);     // pad_amount:   4 DWORDs (=  8 halfs)
    g1[1] = (32 << 16);                               // abar=0 | tensor_dim0 = 32
    g1[2] = (rows_remaining & 0xFFFF) << 16;          // tdim0 hi=0 | tensor_dim1 lo
    g1[3] = ((rows_remaining >> 16) & 0xFFFF)         // tensor_dim1 hi
          | (32 << 16);                               // tile_dim0 = 32
    g1[4] = 32;                                       // tile_dim1 = 32, tile_dim2 = 0
    g1[5] = 256;                                      // tensor_dim0_stride = 256 elems
    g1[6] = 0;
    g1[7] = 0;
    v4i z4 = {0, 0, 0, 0};
#if __clang_major__ >= 23
    v8i z8 = {0, 0, 0, 0, 0, 0, 0, 0};
    __builtin_amdgcn_tensor_load_to_lds(g0, g1, z4, z4, z8, 0);
#else
    __builtin_amdgcn_tensor_load_to_lds(g0, g1, z4, z4, 0);
#endif
}
#endif

// ---------------------------------------------------------------------------
// LSQ quantize-dequantize (forward collapses to round(clip(w/a,-8,7))*a),
// emitted as exact FP16 (4-bit int times scale). alpha==nullptr -> plain cvt.
// ---------------------------------------------------------------------------
__global__ void k_quant(const float* __restrict__ w, const float* __restrict__ alpha,
                        _Float16* __restrict__ out, int n) {
    int i = blockIdx.x * 256 + threadIdx.x;
    if (i >= n) return;
    float v = w[i];
    if (alpha) {
        float a  = alpha[0];
        float wn = fminf(fmaxf(v / a, -8.0f), 7.0f);
        v = rintf(wn) * a;            // round-half-even, matches jnp.round
    }
    out[i] = (_Float16)v;
}

// ---------------------------------------------------------------------------
// elementwise add: c = a + b  (tgt + query_pos)
// ---------------------------------------------------------------------------
__global__ void k_add(const float* __restrict__ a, const float* __restrict__ b,
                      float* __restrict__ c, int n) {
    int i = blockIdx.x * 256 + threadIdx.x;
    if (i < n) c[i] = a[i] + b[i];
}

// ---------------------------------------------------------------------------
// Generic WMMA GEMM: C[M,N] = act(A_f32[M,K] @ W_f16[N,K]^T + bias[N])
// Block tile 128(M) x 64(N), K-step 32. 8 waves, each 32x32 (2x2 WMMA tiles).
// A staged f32->f16 into padded LDS; W fragments read straight from global
// (weights are small & L2-resident). Emits v_wmma_f32_16x16x32_f16.
// ---------------------------------------------------------------------------
#define BK 32
#define LDA 40   // halfs; 80B row stride -> 16B aligned, bank-spread

__global__ __launch_bounds__(256) void k_gemm(
    const float* __restrict__ A, const _Float16* __restrict__ W,
    const float* __restrict__ bias, float* __restrict__ Cf,
    _Float16* __restrict__ Ch, int M, int N, int K, int relu) {
    __shared__ _Float16 sA[128 * LDA];
    const int tid = threadIdx.x, lane = tid & 31, wid = tid >> 5;
    const int l16 = lane & 15, hs = lane >> 4, klo = hs * 8;
    const int bm = blockIdx.x * 128, bn = blockIdx.y * 64;
    const int wm = (wid & 3) * 32, wn = (wid >> 2) * 32;
    v8f acc[2][2] = {};

    for (int k0 = 0; k0 < K; k0 += BK) {
        // stage A tile (128x32 f32 -> f16 LDS), coalesced float4 loads
        for (int it = 0; it < 4; ++it) {
            int idx = tid + it * 256;            // 0..1023
            int r   = idx >> 3;
            int c4  = (idx & 7) * 4;
            int gr  = bm + r; if (gr >= M) gr = M - 1;
            const float* gp = A + (size_t)gr * K + k0 + c4;
            if (k0 + BK < K) __builtin_prefetch(gp + BK, 0, 1);  // global_prefetch
            const float4 av = *(const float4*)gp;
            _Float16* s = sA + r * LDA + c4;
            s[0] = (_Float16)av.x; s[1] = (_Float16)av.y;
            s[2] = (_Float16)av.z; s[3] = (_Float16)av.w;
        }
        __syncthreads();

        for (int sm = 0; sm < 2; ++sm) {
            union { v16h v; uint4 u[2]; } af;
            const _Float16* ap = sA + (wm + sm * 16 + l16) * LDA;
            af.u[0] = *(const uint4*)(ap + klo);        // ds_load_b128
            af.u[1] = *(const uint4*)(ap + 16 + klo);
            for (int sn = 0; sn < 2; ++sn) {
                int n = bn + wn + sn * 16 + l16;
                union { v16h v; uint4 u[2]; } bf;
                const _Float16* bp = W + (size_t)n * K + k0 + klo;
                bf.u[0] = *(const uint4*)(bp);          // global_load_b128
                bf.u[1] = *(const uint4*)(bp + 16);
                acc[sm][sn] = __builtin_amdgcn_wmma_f32_16x16x32_f16(
                    false, af.v, false, bf.v, (short)0, acc[sm][sn], false, false);
            }
        }
        __syncthreads();
    }

    for (int sm = 0; sm < 2; ++sm)
        for (int sn = 0; sn < 2; ++sn)
            for (int r = 0; r < 8; ++r) {
                int row = bm + wm + sm * 16 + hs * 8 + r;
                int col = bn + wn + sn * 16 + l16;
                if (row < M) {
                    float v = acc[sm][sn][r] + (bias ? bias[col] : 0.0f);
                    if (relu) v = fmaxf(v, 0.0f);
                    if (Ch) Ch[(size_t)row * N + col] = (_Float16)v;
                    else    Cf[(size_t)row * N + col] = v;
                }
            }
}

// ---------------------------------------------------------------------------
// Flash self-attention (per block: one (b,h), 128-query tile; 8 waves x 16 q).
// K/V chunks (32 keys x 32 dh) moved to LDS by the Tensor Data Mover (wave 0
// issues two tensor_load_to_lds, waits on TENSORcnt, then block barrier);
// online softmax via width-16 shfl reductions; P transposed C-layout ->
// A-layout through a per-wave LDS scratch (same-wave DS ops are in-order).
// TDM zero-fills out-of-range key rows; scores are masked to -inf anyway.
// ---------------------------------------------------------------------------
__global__ __launch_bounds__(256) void k_flash(
    const _Float16* __restrict__ Q, const _Float16* __restrict__ Kb,
    const _Float16* __restrict__ Vb, float* __restrict__ O) {
    __shared__ __attribute__((aligned(128))) _Float16 sK[32 * 40];
    __shared__ __attribute__((aligned(128))) _Float16 sV[32 * 40];
    __shared__ __attribute__((aligned(128))) _Float16 sP[8 * 16 * 40];
    const int tid = threadIdx.x, lane = tid & 31, wid = tid >> 5;
    const int l16 = lane & 15, hs = lane >> 4, klo = hs * 8;
    const int qt = blockIdx.x * 128;
    const int bh = blockIdx.y, b = bh >> 3, h = bh & 7;
    const size_t ho = (size_t)h * DH_;

    int qrow = qt + wid * 16 + l16; if (qrow >= LQ_) qrow = LQ_ - 1;
    union { v16h v; uint4 u[2]; } qa;
    {
        const _Float16* qp = Q + ((size_t)(b * LQ_ + qrow)) * D_ + ho;
        qa.u[0] = *(const uint4*)(qp + klo);
        qa.u[1] = *(const uint4*)(qp + 16 + klo);
    }
    float mrow[8], lrow[8];
    for (int r = 0; r < 8; ++r) { mrow[r] = -1e30f; lrow[r] = 0.0f; }
    v8f accd[2] = {};
    const float scl = 0.17677669529663687f;   // 1/sqrt(DH)
    _Float16* myP = sP + wid * (16 * 40);

    for (int kc = 0; kc < LQ_; kc += 32) {
#ifdef HAVE_TDM
        if (wid == 0) {   // one wave drives the Tensor Data Mover
            const size_t rowoff = ((size_t)(b * LQ_ + kc)) * D_ + ho;
            tdm_load_tile32(Kb + rowoff, (unsigned)(unsigned long long)(void*)sK, LQ_ - kc);
            tdm_load_tile32(Vb + rowoff, (unsigned)(unsigned long long)(void*)sV, LQ_ - kc);
            __builtin_amdgcn_s_wait_tensorcnt((short)0);
        }
#else
        {   // cooperative staging: threads 0-127 -> K chunk, 128-255 -> V chunk
            int sel = tid >> 7, tt = tid & 127;
            int kr = tt >> 2, c8 = (tt & 3) * 8;
            int key = kc + kr; if (key >= LQ_) key = LQ_ - 1;
            const _Float16* src = (sel ? Vb : Kb) + ((size_t)(b * LQ_ + key)) * D_ + ho + c8;
            _Float16* dst = (sel ? sV : sK) + kr * 40 + c8;
            *(uint4*)dst = *(const uint4*)src;
        }
#endif
        __syncthreads();

        v8f s[2];
        for (int t = 0; t < 2; ++t) {
            union { v16h v; uint4 u[2]; } kf;
            const _Float16* kp = sK + (t * 16 + l16) * 40 + klo;
            kf.u[0] = *(const uint4*)kp;
            kf.u[1] = *(const uint4*)(kp + 16);
            v8f z = {};
            s[t] = __builtin_amdgcn_wmma_f32_16x16x32_f16(
                false, qa.v, false, kf.v, (short)0, z, false, false);
        }

        for (int r = 0; r < 8; ++r) {
            float s0 = s[0][r] * scl, s1 = s[1][r] * scl;
            if (kc + l16      >= LQ_) s0 = -1e30f;
            if (kc + 16 + l16 >= LQ_) s1 = -1e30f;
            float mt = fmaxf(s0, s1);
            for (int m = 1; m < 16; m <<= 1) mt = fmaxf(mt, __shfl_xor(mt, m, 16));
            float mn = fmaxf(mrow[r], mt);
            float p0 = __expf(s0 - mn), p1 = __expf(s1 - mn);
            float rs = p0 + p1;
            for (int m = 1; m < 16; m <<= 1) rs += __shfl_xor(rs, m, 16);
            float al = __expf(mrow[r] - mn);
            lrow[r] = lrow[r] * al + rs;
            mrow[r] = mn;
            accd[0][r] *= al; accd[1][r] *= al;
            int prw = hs * 8 + r;
            myP[prw * 40 + l16]      = (_Float16)p0;
            myP[prw * 40 + 16 + l16] = (_Float16)p1;
        }

        union { v16h v; uint4 u[2]; } pf;     // C-layout -> A-layout via LDS
        const _Float16* pp = myP + l16 * 40;
        pf.u[0] = *(const uint4*)(pp + hs * 16);
        pf.u[1] = *(const uint4*)(pp + hs * 16 + 8);

        for (int dt = 0; dt < 2; ++dt) {
            v16h vf;
            int d = dt * 16 + l16;
            for (int j = 0; j < 16; ++j) vf[j] = sV[(hs * 16 + j) * 40 + d];
            accd[dt] = __builtin_amdgcn_wmma_f32_16x16x32_f16(
                false, pf.v, false, vf, (short)0, accd[dt], false, false);
        }
        __syncthreads();
    }

    for (int dt = 0; dt < 2; ++dt)
        for (int r = 0; r < 8; ++r) {
            int row = qt + wid * 16 + hs * 8 + r;
            if (row < LQ_)
                O[((size_t)(b * LQ_ + row)) * D_ + ho + dt * 16 + l16] = accd[dt][r] / lrow[r];
        }
}

// ---------------------------------------------------------------------------
// Deformable sampling: one wave per (b,q,h), lane = channel (DH=32=wave32).
// Fused softmax over the 16 (level,point) weights + bilinear gather.
// ---------------------------------------------------------------------------
__global__ __launch_bounds__(256) void k_sample(
    const float* __restrict__ ref, const float* __restrict__ offs,
    const float* __restrict__ awr, const _Float16* __restrict__ val,
    float* __restrict__ out) {
    const int HLs[4] = {100, 50, 25, 13};
    const int WLs[4] = {150, 75, 38, 19};
    const int ST[4]  = {0, 15000, 18750, 19700};
    int gw   = blockIdx.x * 8 + (threadIdx.x >> 5);
    int lane = threadIdx.x & 31;
    int h = gw & 7;
    int q = (gw >> 3) % LQ_;
    int b = gw / (LQ_ * H_);
    size_t rowq = (size_t)b * LQ_ + q;

    float w[16]; float mx = -1e30f;
    const float* awp = awr + rowq * 128 + h * 16;
    for (int i = 0; i < 16; ++i) { w[i] = awp[i]; mx = fmaxf(mx, w[i]); }
    float sum = 0.0f;
    for (int i = 0; i < 16; ++i) { w[i] = __expf(w[i] - mx); sum += w[i]; }
    float inv = 1.0f / sum;

    const float* op = offs + rowq * 256 + h * 32;
    float acc = 0.0f;
    for (int l = 0; l < 4; ++l) {
        int Hl = HLs[l], Wl = WLs[l];
        float rx = ref[(rowq * 4 + l) * 2 + 0];
        float ry = ref[(rowq * 4 + l) * 2 + 1];
        const _Float16* vl = val + ((size_t)b * LIN_ + ST[l]) * D_ + h * DH_ + lane;
        for (int p = 0; p < 4; ++p) {
            float x = rx * Wl + op[l * 8 + p * 2 + 0] - 0.5f;   // loc*W - 0.5
            float y = ry * Hl + op[l * 8 + p * 2 + 1] - 0.5f;
            float xf = floorf(x), yf = floorf(y);
            int x0 = (int)xf, y0 = (int)yf;
            float wx1 = x - xf, wy1 = y - yf, wx0 = 1.0f - wx1, wy0 = 1.0f - wy1;
            float smp = 0.0f;
            for (int cy = 0; cy < 2; ++cy)
                for (int cx = 0; cx < 2; ++cx) {
                    int yy = y0 + cy, xx = x0 + cx;
                    if (yy >= 0 && yy < Hl && xx >= 0 && xx < Wl) {
                        float g = (float)vl[((size_t)yy * Wl + xx) * D_];
                        smp += g * ((cy ? wy1 : wy0) * (cx ? wx1 : wx0));
                    }
                }
            acc += w[l * 4 + p] * inv * smp;
        }
    }
    out[rowq * 256 + h * 32 + lane] = acc;
}

// ---------------------------------------------------------------------------
// Residual + LayerNorm: one wave per 256-wide row (8 elems/lane, shfl reduce)
// ---------------------------------------------------------------------------
__global__ __launch_bounds__(256) void k_res_ln(
    const float* __restrict__ res, const float* __restrict__ delta,
    const float* __restrict__ g, const float* __restrict__ beta,
    float* __restrict__ out, int rows) {
    int wid = threadIdx.x >> 5, lane = threadIdx.x & 31;
    int row = blockIdx.x * 8 + wid;
    if (row >= rows) return;
    const float* rp = res + (size_t)row * 256;
    const float* dp = delta + (size_t)row * 256;
    float x[8]; float s = 0.0f;
    for (int j = 0; j < 8; ++j) { int c = lane + j * 32; x[j] = rp[c] + dp[c]; s += x[j]; }
    for (int m = 1; m < 32; m <<= 1) s += __shfl_xor(s, m, 32);
    float mean = s * (1.0f / 256.0f);
    float v = 0.0f;
    for (int j = 0; j < 8; ++j) { float t = x[j] - mean; v += t * t; }
    for (int m = 1; m < 32; m <<= 1) v += __shfl_xor(v, m, 32);
    float iv = rsqrtf(v * (1.0f / 256.0f) + 1e-5f);
    for (int j = 0; j < 8; ++j) {
        int c = lane + j * 32;
        out[(size_t)row * 256 + c] = (x[j] - mean) * iv * g[c] + beta[c];
    }
}

// ---------------------------------------------------------------------------
// Orchestration
// ---------------------------------------------------------------------------
extern "C" void kernel_launch(void* const* d_in, const int* in_sizes, int n_in,
                              void* d_out, int out_size, void* d_ws, size_t ws_size,
                              hipStream_t stream) {
    (void)in_sizes; (void)n_in; (void)out_size; (void)ws_size;
    const float *tgt = (const float*)d_in[0], *qpos = (const float*)d_in[1];
    const float *refp = (const float*)d_in[2], *src = (const float*)d_in[3];
    const float *qW = (const float*)d_in[4],  *qb = (const float*)d_in[5];
    const float *kW = (const float*)d_in[6],  *kb = (const float*)d_in[7];
    const float *vW = (const float*)d_in[8],  *vb = (const float*)d_in[9];
    const float *oW = (const float*)d_in[10], *ob = (const float*)d_in[11];
    const float *a_q = (const float*)d_in[12], *a_k = (const float*)d_in[13];
    const float *a_v = (const float*)d_in[14], *a_o = (const float*)d_in[15];
    const float *val_W = (const float*)d_in[16], *val_b = (const float*)d_in[17];
    const float *a_val = (const float*)d_in[18];
    const float *off_W = (const float*)d_in[19], *off_b = (const float*)d_in[20];
    const float *aw_W  = (const float*)d_in[21], *aw_b  = (const float*)d_in[22];
    const float *out_W = (const float*)d_in[23], *out_b = (const float*)d_in[24];
    const float *a_out = (const float*)d_in[25];
    const float *W1 = (const float*)d_in[26], *b1 = (const float*)d_in[27];
    const float *a_w1 = (const float*)d_in[28];
    const float *W2 = (const float*)d_in[29], *b2 = (const float*)d_in[30];
    const float *a_w2 = (const float*)d_in[31];
    const float *ln1_g = (const float*)d_in[32], *ln1_b = (const float*)d_in[33];
    const float *ln2_g = (const float*)d_in[34], *ln2_b = (const float*)d_in[35];
    const float *ln3_g = (const float*)d_in[36], *ln3_b = (const float*)d_in[37];

    // workspace carving
    char* base = (char*)d_ws; size_t off = 0;
    auto alloc = [&](size_t bytes) -> void* {
        void* r = base + off; off = (off + bytes + 255) & ~(size_t)255; return r;
    };
    const size_t NE = (size_t)NROW * D_;
    const size_t NF = (size_t)NROW * DFF_;
    const size_t NV = (size_t)VROW * D_;
    _Float16 *wq_q  = (_Float16*)alloc(65536 * 2),  *wq_k  = (_Float16*)alloc(65536 * 2);
    _Float16 *wq_v  = (_Float16*)alloc(65536 * 2),  *wq_o  = (_Float16*)alloc(65536 * 2);
    _Float16 *wq_vl = (_Float16*)alloc(65536 * 2),  *wq_ot = (_Float16*)alloc(65536 * 2);
    _Float16 *wq_of = (_Float16*)alloc(65536 * 2),  *wq_aw = (_Float16*)alloc(32768 * 2);
    _Float16 *wq_w1 = (_Float16*)alloc(262144 * 2), *wq_w2 = (_Float16*)alloc(262144 * 2);
    float    *qk_in = (float*)alloc(NE * 4);
    _Float16 *qh = (_Float16*)alloc(NE * 2), *kh = (_Float16*)alloc(NE * 2);
    _Float16 *vh = (_Float16*)alloc(NE * 2);
    float *sa   = (float*)alloc(NE * 4), *sao  = (float*)alloc(NE * 4);
    float *tgt1 = (float*)alloc(NE * 4);
    _Float16 *valh = (_Float16*)alloc(NV * 2);
    float *offs = (float*)alloc(NE * 4);
    float *awr  = (float*)alloc((size_t)NROW * 128 * 4);
    float *accb = (float*)alloc(NE * 4), *ca = (float*)alloc(NE * 4);
    float *tgt2 = (float*)alloc(NE * 4);
    float *ffn1 = (float*)alloc(NF * 4), *ffn2 = (float*)alloc(NE * 4);

    auto quant = [&](const float* w, const float* a, _Float16* o, int n) {
        k_quant<<<(n + 255) / 256, 256, 0, stream>>>(w, a, o, n);
    };
    quant(qW,    a_q,   wq_q,  65536);  quant(kW,   a_k,   wq_k,  65536);
    quant(vW,    a_v,   wq_v,  65536);  quant(oW,   a_o,   wq_o,  65536);
    quant(val_W, a_val, wq_vl, 65536);  quant(out_W, a_out, wq_ot, 65536);
    quant(W1,    a_w1,  wq_w1, 262144); quant(W2,   a_w2,  wq_w2, 262144);
    quant(off_W, nullptr, wq_of, 65536); quant(aw_W, nullptr, wq_aw, 32768);

    auto gemm = [&](const float* A, const _Float16* W, const float* bias,
                    float* Cf, _Float16* Ch, int M, int N, int K, int relu) {
        dim3 grid((M + 127) / 128, N / 64);
        k_gemm<<<grid, 256, 0, stream>>>(A, W, bias, Cf, Ch, M, N, K, relu);
    };

    // ---- self attention ----
    k_add<<<(int)((NE + 255) / 256), 256, 0, stream>>>(tgt, qpos, qk_in, (int)NE);
    gemm(qk_in, wq_q, qb, nullptr, qh, NROW, D_, D_, 0);
    gemm(qk_in, wq_k, kb, nullptr, kh, NROW, D_, D_, 0);
    gemm(tgt,   wq_v, vb, nullptr, vh, NROW, D_, D_, 0);
    {
        dim3 grid((LQ_ + 127) / 128, B_ * H_);
        k_flash<<<grid, 256, 0, stream>>>(qh, kh, vh, sa);
    }
    gemm(sa, wq_o, ob, sao, nullptr, NROW, D_, D_, 0);
    k_res_ln<<<NROW / 8, 256, 0, stream>>>(tgt, sao, ln2_g, ln2_b, tgt1, NROW);

    // ---- deformable cross attention ----
    k_add<<<(int)((NE + 255) / 256), 256, 0, stream>>>(tgt1, qpos, qk_in, (int)NE);
    gemm(src,   wq_vl, val_b, nullptr, valh, VROW, D_, D_, 0);
    gemm(qk_in, wq_of, off_b, offs, nullptr, NROW, D_,  D_, 0);
    gemm(qk_in, wq_aw, aw_b,  awr,  nullptr, NROW, 128, D_, 0);
    k_sample<<<(B_ * LQ_ * H_) / 8, 256, 0, stream>>>(refp, offs, awr, valh, accb);
    gemm(accb, wq_ot, out_b, ca, nullptr, NROW, D_, D_, 0);
    k_res_ln<<<NROW / 8, 256, 0, stream>>>(tgt1, ca, ln1_g, ln1_b, tgt2, NROW);

    // ---- FFN ----
    gemm(tgt2, wq_w1, b1, ffn1, nullptr, NROW, DFF_, D_, 1);
    gemm(ffn1, wq_w2, b2, ffn2, nullptr, NROW, D_, DFF_, 0);
    k_res_ln<<<NROW / 8, 256, 0, stream>>>(tgt2, ffn2, ln3_g, ln3_b, (float*)d_out, NROW);
}
